// CuteInferMLP_82523501626094
// MI455X (gfx1250) — 8-bit matrix operand order repaired and checked by reading, compiled, not run on hardware
//
#include <hip/hip_runtime.h>
#include <math.h>

typedef __attribute__((ext_vector_type(16))) int   v16i;
typedef __attribute__((ext_vector_type(8)))  float v8f;
typedef int v4i_gcc __attribute__((__vector_size__(16)));

#define FP8_MAX 448.0f
#define CHUNK   128
#define BROWS   128
#define BPAD    144

#if defined(__has_builtin)
# if __has_builtin(__builtin_amdgcn_global_load_async_to_lds_b128)
#  define USE_ASYNC_LDS 1
# endif
#endif
#ifndef USE_ASYNC_LDS
# define USE_ASYNC_LDS 0
#endif

__device__ __forceinline__ void wait_async_zero() {
#if USE_ASYNC_LDS
# if defined(__has_builtin)
#  if __has_builtin(__builtin_amdgcn_s_wait_asynccnt)
  __builtin_amdgcn_s_wait_asynccnt(0);
#  else
  asm volatile("s_wait_asynccnt 0x0" ::: "memory");
#  endif
# else
  asm volatile("s_wait_asynccnt 0x0" ::: "memory");
# endif
#endif
}

__device__ __forceinline__ float bf16_to_f32(unsigned short h) {
  return __uint_as_float(((unsigned int)h) << 16);
}
__device__ __forceinline__ unsigned short f32_to_bf16_bits(float x) {
  unsigned int u = __float_as_uint(x);
  u += 0x7FFFu + ((u >> 16) & 1u);
  return (unsigned short)(u >> 16);
}
__device__ __forceinline__ float bf16_round_f32(float x) {
  return bf16_to_f32(f32_to_bf16_bits(x));
}

__device__ __forceinline__ unsigned int f32_to_e4m3(float x) {
  const unsigned int u = __float_as_uint(x);
  const unsigned int sign = (u >> 31) << 7;
  float a = fabsf(x);
  if (a >= 448.0f) return sign | 0x7Eu;
  int e = (int)((u >> 23) & 0xFF) - 127;
  if (e < -6) e = -6;
  float q = rintf(ldexpf(a, 3 - e));
  if (q >= 16.0f) { e += 1; q = 8.0f; }
  unsigned int bits;
  if (e == -6 && q < 8.0f) bits = (unsigned int)q;
  else                     bits = ((unsigned int)(e + 7) << 3) | ((unsigned int)q - 8u);
  if (bits > 0x7Eu) bits = 0x7Eu;
  return sign | bits;
}

__global__ void __launch_bounds__(256)
quant_fp8_kernel(const void* __restrict__ src, int srcIsF32,
                 unsigned char* __restrict__ q, float* __restrict__ scales,
                 int rows, int K)
{
  const int C = K >> 7;
  const long long w = (long long)blockIdx.x * (blockDim.x >> 5) + (threadIdx.x >> 5);
  const long long total = (long long)rows * C;
  if (w >= total) return;
  const int lane = threadIdx.x & 31;
  const long long row = w / C;
  const int c = (int)(w - row * C);
  const long long base = row * (long long)K + (long long)c * CHUNK;

  float v[4];
  if (srcIsF32) {
    const float4 f = *(const float4*)((const float*)src + base + lane * 4);
    v[0] = bf16_round_f32(f.x);  v[1] = bf16_round_f32(f.y);
    v[2] = bf16_round_f32(f.z);  v[3] = bf16_round_f32(f.w);
  } else {
    const uint2 r = *(const uint2*)((const unsigned short*)src + base + lane * 4);
    v[0] = bf16_to_f32((unsigned short)(r.x & 0xFFFFu));
    v[1] = bf16_to_f32((unsigned short)(r.x >> 16));
    v[2] = bf16_to_f32((unsigned short)(r.y & 0xFFFFu));
    v[3] = bf16_to_f32((unsigned short)(r.y >> 16));
  }
  float am = fmaxf(fmaxf(fabsf(v[0]), fabsf(v[1])), fmaxf(fabsf(v[2]), fabsf(v[3])));
#pragma unroll
  for (int m = 1; m < 32; m <<= 1) am = fmaxf(am, __shfl_xor(am, m, 32));
  const float scale = fmaxf(am, 1e-4f) / FP8_MAX;
  const float inv = 1.0f / scale;
  unsigned int packed = 0;
#pragma unroll
  for (int i = 0; i < 4; ++i) packed |= f32_to_e4m3(v[i] * inv) << (8 * i);
  *(unsigned int*)(q + base + lane * 4) = packed;
  if (lane == 0) scales[row * C + c] = scale;
}

__global__ void __launch_bounds__(256)
fp8_gemm_fused(const unsigned char* __restrict__ Aq, const float* __restrict__ Sa,
               const unsigned char* __restrict__ Bq, const float* __restrict__ Sb,
               const unsigned short* __restrict__ bias,
               int M, int N, int K, int epilogue,
               unsigned char* __restrict__ Qout, float* __restrict__ Sout,
               unsigned short* __restrict__ Oout)
{
  const int C  = K >> 7;
  const int CN = N >> 7;
  const int bn = blockIdx.x % CN;
  const int bm = blockIdx.x / CN;
  const int tid = threadIdx.x;
  const int wv = tid >> 5;
  const int lane = tid & 31;
  const int h = lane >> 4, l15 = lane & 15;
  const int mBase = bm * 128 + wv * 16;
  const int nBase = bn * 128;

  __shared__ __align__(16) unsigned char sB[2][BROWS * BPAD];

  auto stage = [&](int c, int buf) {
#pragma unroll
    for (int i = 0; i < 4; ++i) {
      const int idx = tid + 256 * i;
      const int row = idx >> 3, seg = (idx & 7) * 16;
      const unsigned char* gp =
          Bq + (size_t)(nBase + row) * K + (size_t)c * CHUNK + seg;
      unsigned char* lp = &sB[buf][row * BPAD + seg];
#if USE_ASYNC_LDS
      __builtin_amdgcn_global_load_async_to_lds_b128(
          (v4i_gcc*)gp, (v4i_gcc*)lp, 0, 0);
#else
      *(int4*)lp = *(const int4*)gp;
#endif
    }
  };

  float acc[8][8];
#pragma unroll
  for (int t = 0; t < 8; ++t)
#pragma unroll
    for (int v = 0; v < 8; ++v) acc[t][v] = 0.0f;

  const unsigned char* aRow = Aq + (size_t)(mBase + l15) * K + 16 * h;

  stage(0, 0);

  for (int c = 0; c < C; ++c) {
    const int cur = c & 1;

    const unsigned char* ap = aRow + (size_t)c * CHUNK;
    if (c + 1 < C) __builtin_prefetch(ap + CHUNK, 0, 0);
    v16i af;
#pragma unroll
    for (int j = 0; j < 8; ++j) {
      const int* p = (const int*)(ap + 8 * j + 16 * (j >> 1));
      af[2 * j] = p[0]; af[2 * j + 1] = p[1];
    }
    float sxv[8];
#pragma unroll
    for (int v = 0; v < 8; ++v) sxv[v] = Sa[(size_t)(mBase + v + 8 * h) * C + c];

    wait_async_zero();
    __syncthreads();
    if (c + 1 < C) stage(c + 1, 1 - cur);

    const unsigned char* lbase = &sB[cur][l15 * BPAD + 16 * h];
#pragma unroll
    for (int t = 0; t < 8; ++t) {
      const unsigned char* lp = lbase + t * 16 * BPAD;
      v16i bfm;
#pragma unroll
      for (int g = 0; g < 4; ++g) {
        const int* p = (const int*)(lp + 32 * g);
        bfm[4 * g]     = p[0]; bfm[4 * g + 1] = p[1];
        bfm[4 * g + 2] = p[2]; bfm[4 * g + 3] = p[3];
      }
      const float sw = Sb[(size_t)(nBase + t * 16 + l15) * C + c];
      v8f zero = {};
      v8f prod = __builtin_amdgcn_wmma_f32_16x16x128_fp8_fp8(
          af, bfm, (short)0, zero, false, false);
#pragma unroll
      for (int v = 0; v < 8; ++v)
        acc[t][v] = fmaf(prod[v], sxv[v] * sw, acc[t][v]);
    }
  }

  float bcol[8];
#pragma unroll
  for (int t = 0; t < 8; ++t) bcol[t] = bf16_to_f32(bias[nBase + t * 16 + l15]);

  if (epilogue == 0) {
#pragma unroll
    for (int t = 0; t < 8; ++t)
#pragma unroll
      for (int v = 0; v < 8; ++v) {
        const float x = acc[t][v] + bcol[t];
        const float g = 0.5f * x * (1.0f + erff(x * 0.70710678118654752440f));
        acc[t][v] = bf16_round_f32(g);
      }
#pragma unroll
    for (int v = 0; v < 8; ++v) {
      float r = 0.0f;
#pragma unroll
      for (int t = 0; t < 8; ++t) r = fmaxf(r, fabsf(acc[t][v]));
      r = fmaxf(r, __shfl_xor(r, 1, 32));
      r = fmaxf(r, __shfl_xor(r, 2, 32));
      r = fmaxf(r, __shfl_xor(r, 4, 32));
      r = fmaxf(r, __shfl_xor(r, 8, 32));
      const float scale = fmaxf(r, 1e-4f) / FP8_MAX;
      const float inv = 1.0f / scale;
      const int m = mBase + v + 8 * h;
#pragma unroll
      for (int t = 0; t < 8; ++t)
        Qout[(size_t)m * N + nBase + t * 16 + l15] =
            (unsigned char)f32_to_e4m3(acc[t][v] * inv);
      if (l15 == 0) Sout[(size_t)m * CN + bn] = scale;
    }
  } else {
#pragma unroll
    for (int t = 0; t < 8; ++t)
#pragma unroll
      for (int v = 0; v < 8; ++v) {
        const int m = mBase + v + 8 * h;
        Oout[(size_t)m * N + nBase + t * 16 + l15] =
            f32_to_bf16_bits(acc[t][v] + bcol[t]);
      }
  }
  (void)M;
}

extern "C" void kernel_launch(void* const* d_in, const int* in_sizes, int n_in,
                              void* d_out, int out_size, void* d_ws, size_t ws_size,
                              hipStream_t stream) {
  const float*          X0 = (const float*)d_in[0];
  const unsigned short* W0 = (const unsigned short*)d_in[1];
  const unsigned short* b0 = (const unsigned short*)d_in[2];
  const unsigned short* W1 = (const unsigned short*)d_in[3];
  const unsigned short* b1 = (const unsigned short*)d_in[4];
  unsigned short*      out = (unsigned short*)d_out;

  const int Hd = in_sizes[4];
  const int Id = in_sizes[2];
  const int Md = (int)((long long)in_sizes[0] / Hd);
  const int CH = Hd / CHUNK, CI = Id / CHUNK;

  char* ws = (char*)d_ws;
  size_t off = 0;
  auto take = [&](size_t bytes) {
    char* p = ws + off;
    off = (off + bytes + 255) & ~(size_t)255;
    return p;
  };
  unsigned char* Xq  = (unsigned char*)take((size_t)Md * Hd);
  float*         Sx  = (float*)take((size_t)Md * CH * 4);
  unsigned char* W0q = (unsigned char*)take((size_t)Id * Hd);
  float*         Sw0 = (float*)take((size_t)Id * CH * 4);
  unsigned char* W1q = (unsigned char*)take((size_t)Hd * Id);
  float*         Sw1 = (float*)take((size_t)Hd * CI * 4);
  unsigned char* Dq  = (unsigned char*)take((size_t)Md * Id);
  float*         Sd  = (float*)take((size_t)Md * CI * 4);
  (void)ws_size; (void)n_in; (void)out_size;

  const int TPB = 256, WPB = 8;

  {
    const long long waves = (long long)Md * CH;
    quant_fp8_kernel<<<(unsigned)((waves + WPB - 1) / WPB), TPB, 0, stream>>>(
        X0, 1, Xq, Sx, Md, Hd);
  }
  {
    const long long waves = (long long)Id * CH;
    quant_fp8_kernel<<<(unsigned)((waves + WPB - 1) / WPB), TPB, 0, stream>>>(
        W0, 0, W0q, Sw0, Id, Hd);
  }
  {
    const long long waves = (long long)Hd * CI;
    quant_fp8_kernel<<<(unsigned)((waves + WPB - 1) / WPB), TPB, 0, stream>>>(
        W1, 0, W1q, Sw1, Hd, Id);
  }
  {
    const unsigned blocks = (unsigned)((Md / 128) * (Id / 128));
    fp8_gemm_fused<<<blocks, TPB, 0, stream>>>(
        Xq, Sx, W0q, Sw0, b0, Md, Id, Hd,  0, Dq, Sd, nullptr);
  }
  {
    const unsigned blocks = (unsigned)((Md / 128) * (Hd / 128));
    fp8_gemm_fused<<<blocks, TPB, 0, stream>>>(
        Dq, Sd, W1q, Sw1, b1, Md, Hd, Id,  1, nullptr, nullptr, out);
  }
}
